// box_61675730370827
// MI455X (gfx1250) — matrix operand order repaired and checked by reading, not compiled, not run on hardware
//
#include <hip/hip_runtime.h>
#include <hip/hip_bf16.h>

typedef __attribute__((ext_vector_type(16))) _Float16 v16h;
typedef __attribute__((ext_vector_type(8)))  float    v8f;

#define CONF_THRES 0.2f
#define IOU_THRES  0.4f
#define MAX_DET    10
#define TOPK_      5
#define B_IMG      16
#define H_IMG      512
#define W_IMG      512
#define N_BOX      2000
#define N_PAD      2048
#define NMS_THREADS 256

// -------------------------------------------------------------------------
// Kernel 1: per-image sort + greedy NMS + top-10 selection (one block/image)
// -------------------------------------------------------------------------
__global__ __launch_bounds__(NMS_THREADS) void nms_select_kernel(
    const float* __restrict__ rb,    // [B, N_BOX, 5]
    const float* __restrict__ neg,   // [B, MAX_DET, 5]
    float* __restrict__ tb_out,      // [B, MAX_DET, 4]
    float* __restrict__ keep2_out,   // [B, MAX_DET]
    int*   __restrict__ ws)          // [B, MAX_DET, 5] int boxes + keep flag
{
  __shared__ float s_score[N_PAD];
  __shared__ int   s_idx[N_PAD];
  __shared__ float s_x1[N_BOX], s_y1[N_BOX], s_x2[N_BOX], s_y2[N_BOX];
  __shared__ unsigned kA[64], kB[64];
  __shared__ float s_comb[MAX_DET * 5];
  __shared__ float s_cb[MAX_DET * 5];

  const int b   = blockIdx.x;
  const int tid = threadIdx.x;
  const float* rbb = rb + (size_t)b * N_BOX * 5;

  if (tid == 0) __builtin_prefetch(rbb, 0, 1);   // global_prefetch_b8 into L2

  // --- load scores, pad to 2048 with -inf ---
  for (int n = tid; n < N_PAD; n += NMS_THREADS) {
    s_score[n] = (n < N_BOX) ? rbb[n * 5 + 4] : -3.0e38f;
    s_idx[n]   = n;
  }

  // --- bitonic sort, descending by score (argsort(-scores)) ---
  for (int k = 2; k <= N_PAD; k <<= 1) {
    for (int j = k >> 1; j > 0; j >>= 1) {
      __syncthreads();
      for (int i = tid; i < N_PAD; i += NMS_THREADS) {
        int ixj = i ^ j;
        if (ixj > i) {
          bool desc = ((i & k) == 0);
          float a = s_score[i], c = s_score[ixj];
          if (desc ? (a < c) : (a > c)) {
            s_score[i] = c; s_score[ixj] = a;
            int t = s_idx[i]; s_idx[i] = s_idx[ixj]; s_idx[ixj] = t;
          }
        }
      }
    }
  }
  __syncthreads();

  // --- gather sorted boxes into SoA LDS ---
  for (int n = tid; n < N_BOX; n += NMS_THREADS) {
    int src = s_idx[n];
    s_x1[n] = rbb[src * 5 + 0];
    s_y1[n] = rbb[src * 5 + 1];
    s_x2[n] = rbb[src * 5 + 2];
    s_y2[n] = rbb[src * 5 + 3];
  }
  // --- init keep bitmasks from sorted confidences ---
  if (tid < 64) {
    unsigned wa = 0u, wb = 0u;
    for (int t2 = 0; t2 < 32; ++t2) {
      int n = (tid << 5) + t2;
      if (n < N_BOX) {
        float c = s_score[n];
        if (c > CONF_THRES) wa |= 1u << t2;
        if (c > 0.0f)       wb |= 1u << t2;
      }
    }
    kA[tid] = wa; kB[tid] = wb;
  }
  __syncthreads();

  // --- register-resident candidate boxes: 8 per lane (stride 256) ---
  float jx1[8], jy1[8], jx2[8], jy2[8], jar[8];
#pragma unroll
  for (int s = 0; s < 8; ++s) {
    int j = tid + (s << 8);
    if (j < N_BOX) {
      jx1[s] = s_x1[j]; jy1[s] = s_y1[j];
      jx2[s] = s_x2[j]; jy2[s] = s_y2[j];
      jar[s] = fmaxf(jx2[s] - jx1[s], 0.f) * fmaxf(jy2[s] - jy1[s], 0.f);
    } else {
      jx1[s] = 0.f; jy1[s] = 0.f; jx2[s] = 0.f; jy2[s] = 0.f; jar[s] = 0.f;
    }
  }

  // --- sequential greedy NMS, dual masks; barrier only when we suppressed ---
  for (int i = 0; i < N_BOX - 1; ++i) {
    unsigned wA = kA[i >> 5], wB = kB[i >> 5];   // uniform broadcast LDS read
    bool ka = (wA >> (i & 31)) & 1u;
    bool kb = (wB >> (i & 31)) & 1u;
    if (ka | kb) {                                // uniform branch
      float ix1 = s_x1[i], iy1 = s_y1[i], ix2 = s_x2[i], iy2 = s_y2[i];
      float ia = fmaxf(ix2 - ix1, 0.f) * fmaxf(iy2 - iy1, 0.f);
#pragma unroll
      for (int s = 0; s < 8; ++s) {
        int j = tid + (s << 8);
        if (j > i && j < N_BOX) {
          float xx1 = fmaxf(ix1, jx1[s]);
          float yy1 = fmaxf(iy1, jy1[s]);
          float xx2 = fminf(ix2, jx2[s]);
          float yy2 = fminf(iy2, jy2[s]);
          float inter = fmaxf(xx2 - xx1, 0.f) * fmaxf(yy2 - yy1, 0.f);
          float iou = inter / (ia + jar[s] - inter + 1e-9f);
          if (iou > IOU_THRES) {
            unsigned m = ~(1u << (j & 31));
            if (ka) atomicAnd(&kA[j >> 5], m);    // ds_and_b32
            if (kb) atomicAnd(&kB[j >> 5], m);
          }
        }
      }
      __syncthreads();
    }
  }
  __syncthreads();

  // --- tiny-box filter on both masks ---
  for (int n = tid; n < N_BOX; n += NMS_THREADS) {
    bool ok = (s_x2[n] - s_x1[n] >= 1.0f) && (s_y2[n] - s_y1[n] >= 1.0f);
    if (!ok) {
      unsigned m = ~(1u << (n & 31));
      atomicAnd(&kA[n >> 5], m);
      atomicAnd(&kB[n >> 5], m);
    }
  }
  __syncthreads();

  // --- small sequential tail on thread 0 ---
  if (tid == 0) {
    // keepB limited to first TOPK kept (cumsum rule)
    int cnt = 0;
    for (int n = 0; n < N_BOX; ++n) {
      unsigned m = 1u << (n & 31);
      if (kB[n >> 5] & m) {
        if (cnt >= TOPK_) kB[n >> 5] &= ~m; else ++cnt;
      }
    }
    bool has_conf = s_score[0] > CONF_THRES;      // max of sorted scores
    const unsigned* keep = has_conf ? kA : kB;
    const float* negb = neg + (size_t)b * MAX_DET * 5;

    // first min(count,10) kept boxes in sorted order, neg backfill
    int nk = 0;
    for (int n = 0; n < N_BOX && nk < MAX_DET; ++n) {
      if (keep[n >> 5] & (1u << (n & 31))) {
        s_comb[nk * 5 + 0] = s_x1[n];
        s_comb[nk * 5 + 1] = s_y1[n];
        s_comb[nk * 5 + 2] = s_x2[n];
        s_comb[nk * 5 + 3] = s_y2[n];
        s_comb[nk * 5 + 4] = s_score[n];
        ++nk;
      }
    }
    for (int s = nk; s < MAX_DET; ++s)
      for (int c = 0; c < 5; ++c) s_comb[s * 5 + c] = negb[s * 5 + c];

    // stable descending sort of 10 by conf (rank counting)
    for (int i = 0; i < MAX_DET; ++i) {
      float ci = s_comb[i * 5 + 4];
      int pos = 0;
      for (int j2 = 0; j2 < MAX_DET; ++j2) {
        float cj = s_comb[j2 * 5 + 4];
        if (cj > ci || (cj == ci && j2 < i)) ++pos;
      }
      for (int c = 0; c < 5; ++c) s_cb[pos * 5 + c] = s_comb[i * 5 + c];
    }

    // greedy NMS over the 10 (valid = conf > 0)
    unsigned k2 = 0;
    for (int i = 0; i < MAX_DET; ++i)
      if (s_cb[i * 5 + 4] > 0.0f) k2 |= 1u << i;
    for (int i = 0; i < MAX_DET; ++i) {
      if (!((k2 >> i) & 1u)) continue;
      float x1 = s_cb[i*5+0], y1 = s_cb[i*5+1], x2 = s_cb[i*5+2], y2 = s_cb[i*5+3];
      float ia = fmaxf(x2 - x1, 0.f) * fmaxf(y2 - y1, 0.f);
      for (int j2 = i + 1; j2 < MAX_DET; ++j2) {
        float ja = fmaxf(s_cb[j2*5+2] - s_cb[j2*5+0], 0.f) *
                   fmaxf(s_cb[j2*5+3] - s_cb[j2*5+1], 0.f);
        float xx1 = fmaxf(x1, s_cb[j2*5+0]);
        float yy1 = fmaxf(y1, s_cb[j2*5+1]);
        float xx2 = fminf(x2, s_cb[j2*5+2]);
        float yy2 = fminf(y2, s_cb[j2*5+3]);
        float inter = fmaxf(xx2 - xx1, 0.f) * fmaxf(yy2 - yy1, 0.f);
        float iou = inter / (ia + ja - inter + 1e-9f);
        if (iou > IOU_THRES) k2 &= ~(1u << j2);
      }
    }

    // write target_boxes, keep2, and rasterizer workspace
    float* tb = tb_out + (size_t)b * MAX_DET * 4;
    float* kk = keep2_out + (size_t)b * MAX_DET;
    int* wsb = ws + b * MAX_DET * 5;
    for (int s = 0; s < MAX_DET; ++s) {
      bool kp = (k2 >> s) & 1u;
      for (int c = 0; c < 4; ++c) tb[s * 4 + c] = kp ? s_cb[s * 5 + c] : 0.0f;
      kk[s] = kp ? 1.0f : 0.0f;
      wsb[s * 5 + 0] = (int)floorf(s_cb[s * 5 + 0] + 0.5f);
      wsb[s * 5 + 1] = (int)floorf(s_cb[s * 5 + 1] + 0.5f);
      wsb[s * 5 + 2] = (int)floorf(s_cb[s * 5 + 2] + 0.5f);
      wsb[s * 5 + 3] = (int)floorf(s_cb[s * 5 + 3] + 0.5f);
      wsb[s * 5 + 4] = kp ? 1 : 0;
    }
  }
}

// -------------------------------------------------------------------------
// Kernel 2: coverage mask as (512 x K) x (K x 512) boolean matmul via WMMA.
// One wave = one 16x16 output tile; K=10 padded to 32 (f16, exact).
// -------------------------------------------------------------------------
__global__ __launch_bounds__(256) void raster_wmma_kernel(
    const int* __restrict__ ws, float* __restrict__ mask_out)
{
  const int lane = threadIdx.x & 31;
  const int wv   = threadIdx.x >> 5;
  int g  = blockIdx.x * 8 + wv;         // global tile id, 1024 tiles/image
  int b  = g >> 10;
  int t  = g & 1023;
  int y0 = (t >> 5) << 4;
  int x0 = (t & 31) << 4;

  const int* wsb = ws + b * MAX_DET * 5;
  int bx1[MAX_DET], by1[MAX_DET], bx2[MAX_DET], by2[MAX_DET], bk[MAX_DET];
#pragma unroll
  for (int k = 0; k < MAX_DET; ++k) {
    bx1[k] = wsb[k * 5 + 0];
    by1[k] = wsb[k * 5 + 1];
    bx2[k] = wsb[k * 5 + 2];
    by2[k] = wsb[k * 5 + 3];
    bk[k]  = wsb[k * 5 + 4];
  }

  const int halfSel = (lane >> 4) & 1;

  // A fragment (16x32 f16): A[y,k] = keep2[k] && y in [y1,y2)
  int mrow = y0 + (lane & 15);
  v16h a;
#pragma unroll
  for (int e = 0; e < 16; ++e) {
    int K = e + (halfSel << 4);
    bool v = (K < MAX_DET) && bk[K] && (mrow >= by1[K]) && (mrow < by2[K]);
    a[e] = v ? (_Float16)1.0f : (_Float16)0.0f;
  }

  // B fragment (32x16 f16): B[k,x] = x in [x1,x2)
  int ncol = x0 + (lane & 15);
  v16h bm;
#pragma unroll
  for (int e = 0; e < 16; ++e) {
    int K = e + (halfSel << 4);
    bool v = (K < MAX_DET) && (ncol >= bx1[K]) && (ncol < bx2[K]);
    bm[e] = v ? (_Float16)1.0f : (_Float16)0.0f;
  }

  v8f c = {};
  // 8 args: (neg_a, A, neg_b, B, c_mod, C, reuse_a, reuse_b)
  c = __builtin_amdgcn_wmma_f32_16x16x32_f16(false, a, false, bm,
                                             (short)0, c, false, false);

  // mask = ~covered : covered iff sum > 0
  float* mb = mask_out + (size_t)b * (H_IMG * W_IMG);
#pragma unroll
  for (int r = 0; r < 8; ++r) {
    int row = y0 + r + (halfSel << 3);
    mb[row * W_IMG + ncol] = (c[r] > 0.5f) ? 0.0f : 1.0f;
  }
}

// -------------------------------------------------------------------------
extern "C" void kernel_launch(void* const* d_in, const int* in_sizes, int n_in,
                              void* d_out, int out_size, void* d_ws, size_t ws_size,
                              hipStream_t stream) {
  // inputs: d_in[0]=x (unused by reference), d_in[1]=region_boxes, d_in[2]=neg_boxes
  const float* rb  = (const float*)d_in[1];
  const float* neg = (const float*)d_in[2];

  float* out      = (float*)d_out;
  float* mask_out = out;                                        // [16,512,512]
  float* tb_out   = out + (size_t)B_IMG * H_IMG * W_IMG;        // [16,10,4]
  float* k2_out   = tb_out + B_IMG * MAX_DET * 4;               // [16,10]
  int*   ws       = (int*)d_ws;                                 // [16,10,5]

  nms_select_kernel<<<B_IMG, NMS_THREADS, 0, stream>>>(rb, neg, tb_out, k2_out, ws);
  raster_wmma_kernel<<<(B_IMG * 1024) / 8, 256, 0, stream>>>(ws, mask_out);
}
